// LiquidLayer_53231824667271
// MI455X (gfx1250) — matrix operand order repaired and checked by reading, not compiled, not run on hardware
//
#include <hip/hip_runtime.h>
#include <hip/hip_bf16.h>
#include <math.h>

// Problem constants (from the reference): B=64, T=512, F=256, H=512, dt=1.
constexpr int BB = 64;
constexpr int TT = 512;
constexpr int FF = 256;
constexpr int HH = 512;
constexpr int MM = BB * TT; // 32768 rows of the xproj GEMM

typedef __attribute__((ext_vector_type(16))) __bf16 v16bf;
typedef __attribute__((ext_vector_type(8)))  float  v8f;

// ---------------------------------------------------------------------------
// Kernel 0: swizzle Wh (f32 row-major [n][k]; B[k][n] = Wh[n][k]) into bf16
// B-fragment order: idx = ((ktile*32 + ntile)*32 + lane)*16 + elem
//   lane -> N column (lane&15); lane<16 -> elems K k0+0..7,16..23, lane>=16 -> K k0+8..15,24..31
// ---------------------------------------------------------------------------
__global__ void swizzle_wh_kernel(const float* __restrict__ Wh,
                                  __bf16* __restrict__ whs) {
  int idx   = blockIdx.x * blockDim.x + threadIdx.x;   // < 512*512
  int e     = idx & 15;
  int lane  = (idx >> 4) & 31;
  int ntile = (idx >> 9) & 31;
  int ktile = idx >> 14;
  int k = ktile * 32 + ((lane < 16) ? 0 : 8) + (e < 8 ? e : e + 8);
  int n = ntile * 16 + (lane & 15);
  whs[idx] = (__bf16)Wh[(size_t)n * HH + k];
}

// ---------------------------------------------------------------------------
// Kernel 1: xproj[m][n] = sum_f x[m][f]*Wx[n][f] + bias[n], staged into out.
// Each wave computes 4 adjacent 16x16 N-tiles sharing one A fragment:
// per k-step 1 A build + 4 B builds + 4 WMMAs (4x less x traffic/cvt).
// ---------------------------------------------------------------------------
__global__ __launch_bounds__(256) void xproj_kernel(
    const float* __restrict__ x, const float* __restrict__ Wx,
    const float* __restrict__ bias, float* __restrict__ out) {
  const int lane  = threadIdx.x & 31;
  const int wid   = blockIdx.x * (blockDim.x >> 5) + (threadIdx.x >> 5);
  const int ngrp  = wid & 7;           // 8 groups of 4 N-tiles (H/16/4)
  const int mtile = wid >> 3;          // M/16 = 2048 row tiles
  const int row   = lane & 15;
  const int m     = mtile * 16 + row;      // A row for this lane
  const int kbase = (lane < 16) ? 0 : 8;   // A elem->K mapping
  const int kboff = (lane < 16) ? 0 : 16;  // unused; B uses kbase like A

  const float* __restrict__ xrow = x + (size_t)m * FF;

  v8f acc[4] = {v8f{}, v8f{}, v8f{}, v8f{}};
  #pragma unroll
  for (int k0 = 0; k0 < FF; k0 += 32) {
    v16bf a;
    #pragma unroll
    for (int i = 0; i < 8; ++i) {
      a[i]     = (__bf16)xrow[k0 + kbase + i];
      a[i + 8] = (__bf16)xrow[k0 + 16 + kbase + i];
    }
    #pragma unroll
    for (int s = 0; s < 4; ++s) {
      const int n = (ngrp * 4 + s) * 16 + row;   // B/C column for this lane
      const float* __restrict__ wrow = Wx + (size_t)n * FF;
      v16bf b;
      #pragma unroll
      for (int e = 0; e < 16; ++e) b[e] = (__bf16)wrow[k0 + kbase + (e < 8 ? e : e + 8)];
      acc[s] = __builtin_amdgcn_wmma_f32_16x16x32_bf16(false, a, false, b,
                                                       (short)0, acc[s], false, false);
    }
  }
  #pragma unroll
  for (int s = 0; s < 4; ++s) {
    const int n  = (ngrp * 4 + s) * 16 + row;
    const float bn = bias[n];
    #pragma unroll
    for (int j = 0; j < 8; ++j) {
      const int mj = mtile * 16 + j + ((lane < 16) ? 0 : 8);
      out[(size_t)mj * HH + n] = acc[s][j] + bn;
    }
  }
}

// ---------------------------------------------------------------------------
// Kernel 2: persistent recurrent scan. 4 workgroups (16 batch rows each),
// 1024 threads = 32 waves; wave w owns h columns [16w, 16w+16).
// Wh B-fragments are loop-invariant -> hoisted on-chip once:
//   k-tiles 0..7  -> registers (64 VGPRs/lane)
//   k-tiles 8..15 -> LDS stage (256 KB) ; + 16 KB bf16 h state = 272 KB LDS
// Steady state: WMMA + LDS only; xp read / h write hit L2-resident `out`.
// ---------------------------------------------------------------------------
__global__ __launch_bounds__(1024, 1) void liquid_scan_kernel(
    const float* __restrict__ log_tau, const __bf16* __restrict__ whs,
    const int* __restrict__ steps_p, float* __restrict__ out) {
  extern __shared__ __align__(32) char smem[];
  __bf16* __restrict__ hbuf = (__bf16*)smem;                       // 16*512 bf16
  __bf16* __restrict__ bst  = (__bf16*)(smem + 16 * HH * 2);       // 32*8*512 bf16

  const int lane   = threadIdx.x & 31;
  const int wave   = threadIdx.x >> 5;      // 0..31 -> column tile
  const int bm     = blockIdx.x * 16;       // base batch row
  const int n      = wave * 16 + (lane & 15);
  const int rowsel = (lane < 16) ? 0 : 8;   // C/D: lane half -> M offset
  const int kbase  = (lane < 16) ? 0 : 8;   // A elem->K mapping
  const int arow   = lane & 15;             // A row in LDS h

  const float gain = expf(-log_tau[n]);     // dt/tau with dt=1
  int ne = steps_p[0];
  if (ne < 1) ne = 1;

  // Hoist loop-invariant B fragments: 8 k-tiles to registers.
  v16bf breg[8];
  #pragma unroll
  for (int kt = 0; kt < 8; ++kt)
    breg[kt] = *reinterpret_cast<const v16bf*>(
        whs + ((size_t)(kt * 32 + wave) * 32 + lane) * 16);
  // 8 k-tiles to LDS (each wave stages its own column block: 8 KB).
  #pragma unroll
  for (int kt = 0; kt < 8; ++kt) {
    const v16bf v = *reinterpret_cast<const v16bf*>(
        whs + ((size_t)((kt + 8) * 32 + wave) * 32 + lane) * 16);
    *reinterpret_cast<v16bf*>(bst + ((wave * 8 + kt) * 32 + lane) * 16) = v;
  }
  for (int i = threadIdx.x; i < 16 * HH; i += blockDim.x) hbuf[i] = (__bf16)0.0f;
  __syncthreads();

  for (int t = 0; t < TT; ++t) {
    float xp[8], hn[8];
    #pragma unroll
    for (int j = 0; j < 8; ++j)
      xp[j] = out[(size_t)(bm + j + rowsel) * (TT * HH) + (size_t)t * HH + n];
    if (t + 1 < TT) {  // prefetch next timestep's xp rows (global_prefetch_b8)
      #pragma unroll
      for (int j = 0; j < 8; ++j)
        __builtin_prefetch(
            &out[(size_t)(bm + j + rowsel) * (TT * HH) + (size_t)(t + 1) * HH + n], 0, 1);
    }

    for (int e = 0; e < ne; ++e) {
      v8f acc = {};
      #pragma unroll
      for (int kt = 0; kt < HH / 32; ++kt) {
        const int k0 = kt * 32;
        v16bf a;
        const __bf16* __restrict__ hp = &hbuf[arow * HH + k0 + kbase];
        #pragma unroll
        for (int i = 0; i < 8; ++i) { a[i] = hp[i]; a[i + 8] = hp[16 + i]; }
        const v16bf b = (kt < 8)
            ? breg[kt]
            : *reinterpret_cast<const v16bf*>(
                  bst + ((wave * 8 + (kt - 8)) * 32 + lane) * 16);
        acc = __builtin_amdgcn_wmma_f32_16x16x32_bf16(false, a, false, b,
                                                      (short)0, acc, false, false);
      }
      __syncthreads();  // all waves done reading h before anyone rewrites it
      #pragma unroll
      for (int j = 0; j < 8; ++j) {
        const float z  = tanhf(acc[j] + xp[j]);
        const float ho = (float)hbuf[(j + rowsel) * HH + n];
        hn[j] = ho + (z - ho) * gain;
        hbuf[(j + rowsel) * HH + n] = (__bf16)hn[j];
      }
      __syncthreads();  // new h visible to all waves
    }
    #pragma unroll
    for (int j = 0; j < 8; ++j)
      out[(size_t)(bm + j + rowsel) * (TT * HH) + (size_t)t * HH + n] = hn[j];
  }
}

// ---------------------------------------------------------------------------
extern "C" void kernel_launch(void* const* d_in, const int* in_sizes, int n_in,
                              void* d_out, int out_size, void* d_ws, size_t ws_size,
                              hipStream_t stream) {
  const float* x     = (const float*)d_in[0];
  const float* Wx_w  = (const float*)d_in[1];
  const float* Wh_w  = (const float*)d_in[2];
  const float* bias  = (const float*)d_in[3];
  const float* ltau  = (const float*)d_in[4];
  const int*   steps = (const int*)d_in[5];
  float* out = (float*)d_out;
  __bf16* whs = (__bf16*)d_ws;  // 512*512 bf16 = 512 KB scratch

  // 0) swizzle Wh into WMMA B-fragment order (bf16)
  swizzle_wh_kernel<<<(HH * HH) / 256, 256, 0, stream>>>(Wh_w, whs);
  // 1) xproj GEMM staged into d_out (consumed in place by the scan)
  xproj_kernel<<<(MM / 16) * (HH / 16) / (8 * 4), 256, 0, stream>>>(x, Wx_w, bias, out);
  // 2) persistent recurrent scan, 4 independent batch-slices
  const size_t smem_bytes = (size_t)16 * HH * 2      // bf16 h state (16 KB)
                          + (size_t)32 * 8 * 32 * 16 * 2;  // staged B half (256 KB)
  liquid_scan_kernel<<<BB / 16, 1024, smem_bytes, stream>>>(ltau, whs, steps, out);
}
